// DynamicTransformerBlock_77412490543646
// MI455X (gfx1250) — matrix operand order repaired and checked by reading, not compiled, not run on hardware
//
#include <hip/hip_runtime.h>
#include <hip/hip_bf16.h>

// ---------------- constants ----------------
constexpr int nB = 2, nS = 2048, nSC = 512, nDIM = 2560, nH = 20, nHD = 128, nFFN = 10240;
constexpr float kEPS = 1e-06f;
constexpr float kAttnScale = 0.08838834764831845f; // 128^-0.5

typedef unsigned short u16;
typedef __attribute__((ext_vector_type(16))) __bf16 v16bf;
typedef __attribute__((ext_vector_type(8)))  __bf16 v8bf;
typedef __attribute__((ext_vector_type(8)))  float  v8f;
typedef __attribute__((ext_vector_type(4)))  unsigned int v4u;
typedef int v4i __attribute__((vector_size(16)));

union Frag16 { v16bf v; v8bf h[2]; };

// ---- gfx1250 async global->LDS path (ASYNCcnt), with sync fallback ----
#if defined(__has_builtin)
#if __has_builtin(__builtin_amdgcn_global_load_async_to_lds_b128) && \
    __has_builtin(__builtin_amdgcn_s_wait_asynccnt)
#define USE_ASYNC_LDS 1
#endif
#endif
#ifndef USE_ASYNC_LDS
#define USE_ASYNC_LDS 0
#endif

__device__ __forceinline__ void async_copy16(const u16* g, u16* l) {
#if USE_ASYNC_LDS
    __builtin_amdgcn_global_load_async_to_lds_b128(
        (__attribute__((address_space(1))) v4i*)(void*)g,
        (__attribute__((address_space(3))) v4i*)(void*)l, 0, 0);
#else
    *(v4u*)l = *(const v4u*)g;
#endif
}

__device__ __forceinline__ void async_wait_all() {
#if USE_ASYNC_LDS
    __builtin_amdgcn_s_wait_asynccnt(0);
#endif
}

__device__ __forceinline__ u16 f2bf(float f) {
    unsigned int u = __float_as_uint(f);
    unsigned int r = (u + 0x7FFFu + ((u >> 16) & 1u)) >> 16;
    return (u16)r;
}

__device__ __forceinline__ v8f wmma_bf16(v16bf a, v16bf b, v8f c) {
    // D = A(16x32 bf16) * B(32x16 bf16) + C(16x16 f32)
    return __builtin_amdgcn_wmma_f32_16x16x32_bf16(false, a, false, b, (short)0, c, false, false);
}

__device__ __forceinline__ float gelu_tanh(float x) {
    float x3 = x * x * x;
    return 0.5f * x * (1.0f + tanhf(0.7978845608028654f * (x + 0.044715f * x3)));
}

__device__ __forceinline__ float blk_sum(float v) {
    __shared__ float sred[256];
    int t = threadIdx.x;
    sred[t] = v; __syncthreads();
    #pragma unroll
    for (int s = 128; s > 0; s >>= 1) { if (t < s) sred[t] += sred[t + s]; __syncthreads(); }
    float r = sred[0]; __syncthreads();
    return r;
}

// ---------------- elementwise kernels ----------------

// mod[b,i,d] = sst[i,d] + temb[b,i,d]
__global__ __launch_bounds__(256) void k_mod(const float* __restrict__ sst,
                                             const float* __restrict__ temb,
                                             float* __restrict__ mod) {
    long i = (long)blockIdx.x * 256 + threadIdx.x;
    long n = (long)nB * 6 * nDIM;
    if (i < n) {
        long id = i % ((long)6 * nDIM);   // i,d combined
        mod[i] = sst[id] + temb[i];
    }
}

__global__ __launch_bounds__(256) void k_cvt(const float* __restrict__ src,
                                             u16* __restrict__ dst, long n) {
    long i = (long)blockIdx.x * 256 + threadIdx.x;
    if (i < n) dst[i] = f2bf(src[i]);
}

// mode 0: out = ln(x)*(1+mod[b,scale_idx]) + mod[b,shift_idx]
// mode 1: out = ln(x)*gamma + beta
__global__ __launch_bounds__(256) void k_ln(const float* __restrict__ x,
                                            u16* __restrict__ out,
                                            const float* __restrict__ mod,
                                            int shift_idx, int scale_idx,
                                            const float* __restrict__ gamma,
                                            const float* __restrict__ beta,
                                            int rows_per_batch, int mode) {
    long row = blockIdx.x;
    const float* xr = x + row * nDIM;
    float s = 0.f, s2 = 0.f;
    for (int d = threadIdx.x; d < nDIM; d += 256) { float v = xr[d]; s += v; s2 += v * v; }
    float sum  = blk_sum(s);
    float sum2 = blk_sum(s2);
    float mean = sum * (1.0f / nDIM);
    float var  = sum2 * (1.0f / nDIM) - mean * mean;
    float rstd = rsqrtf(var + kEPS);
    int b = (int)(row / rows_per_batch);
    const float* shp = mod ? (mod + ((long)b * 6 + shift_idx) * nDIM) : nullptr;
    const float* scp = mod ? (mod + ((long)b * 6 + scale_idx) * nDIM) : nullptr;
    for (int d = threadIdx.x; d < nDIM; d += 256) {
        float y = (xr[d] - mean) * rstd;
        if (mode == 0) y = y * (1.0f + scp[d]) + shp[d];
        else           y = y * gamma[d] + beta[d];
        out[row * nDIM + d] = f2bf(y);
    }
}

// RMS over full DIM row, optional RoPE, write bf16 into [B*H, seq, HD]
__global__ __launch_bounds__(256) void k_rmsrope(const float* __restrict__ src, long row_stride,
                                                 const float* __restrict__ w,
                                                 const float* __restrict__ cosb,
                                                 const float* __restrict__ sinb,
                                                 u16* __restrict__ dst, int seq, int do_rope) {
    long row = blockIdx.x;                   // b*seq + s
    int b = (int)(row / seq), s = (int)(row % seq);
    const float* xr = src + row * row_stride;
    float ss = 0.f;
    for (int d = threadIdx.x; d < nDIM; d += 256) { float v = xr[d]; ss += v * v; }
    float sum = blk_sum(ss);
    float r = rsqrtf(sum * (1.0f / nDIM) + kEPS);
    for (int d = threadIdx.x; d < nDIM; d += 256) {
        int hh = d >> 7, hd = d & 127;
        float y = xr[d] * r * w[d];
        float o;
        if (do_rope) {
            int pd = (hd < 64) ? d + 64 : d - 64;
            float yp = xr[pd] * r * w[pd];
            float rot = (hd < 64) ? -yp : yp;
            o = y * cosb[(long)s * nHD + hd] + rot * sinb[(long)s * nHD + hd];
        } else {
            o = y;
        }
        dst[(((long)b * nH + hh) * seq + s) * nHD + hd] = f2bf(o);
    }
}

// pack fp32 [rows, row_stride] (col offset pre-applied) into bf16 heads [B*H, seq, HD]
__global__ __launch_bounds__(256) void k_pack(const float* __restrict__ src, long row_stride,
                                              u16* __restrict__ dst, int seq) {
    long row = blockIdx.x;
    int b = (int)(row / seq), s = (int)(row % seq);
    const float* xr = src + row * row_stride;
    for (int d = threadIdx.x; d < nDIM; d += 256) {
        dst[(((long)b * nH + (d >> 7)) * seq + s) * nHD + (d & 127)] = f2bf(xr[d]);
    }
}

// ---------------- WMMA GEMM: C[M,N] = A[M,K](bf16) * W[N,K]^T(bf16) + bias ----------------
// Double-buffered LDS with async global->LDS copies.
// EPI 0: f32 out    1: bf16 gelu out    2: f32 res+out    3: f32 res+gate*out
template<int EPI>
__global__ __launch_bounds__(256) void k_gemm(const u16* __restrict__ A,
                                              const u16* __restrict__ W,
                                              const float* __restrict__ bias,
                                              void* __restrict__ outv,
                                              const float* __restrict__ res,
                                              const float* __restrict__ modbuf,
                                              int gate_idx, int rows_per_batch,
                                              int M, int N, int K) {
    constexpr int BM = 128, BN = 128, BK = 32, LDT = BK + 8; // padded LDS stride
    __shared__ u16 As[2][BM * LDT];
    __shared__ u16 Bs[2][BN * LDT];

    const int tid = threadIdx.x;
    const int wave = tid >> 5, lane = tid & 31;
    const int hl = lane >> 4, lo16 = lane & 15;
    const int waveM = wave >> 2;          // 0..1 -> 64 rows each
    const int waveN = wave & 3;           // 0..3 -> 32 cols each
    const long row0 = (long)blockIdx.y * BM;
    const long col0 = (long)blockIdx.x * BN;

    v8f zero = {};
    v8f acc[4][2];
    #pragma unroll
    for (int i = 0; i < 4; ++i)
        #pragma unroll
        for (int j = 0; j < 2; ++j) acc[i][j] = zero;

    // per-thread copy coordinates: 512 chunks of 8 bf16 per tile
    auto load_tile = [&](int buf, int kt) {
        #pragma unroll
        for (int i = 0; i < 2; ++i) {
            int cid = tid + 256 * i;
            int r = cid >> 2, c = (cid & 3) * 8;
            async_copy16(&A[(row0 + r) * (long)K + kt + c], &As[buf][r * LDT + c]);
            async_copy16(&W[(col0 + r) * (long)K + kt + c], &Bs[buf][r * LDT + c]);
        }
    };

    const int nk = K / BK;
    load_tile(0, 0);

    for (int t = 0; t < nk; ++t) {
        const int buf = t & 1;
        async_wait_all();
        __syncthreads();                  // tile t resident; prior reads of buf^1 done
        if (t + 1 < nk) {
            load_tile(buf ^ 1, (t + 1) * BK);
            int r = tid >> 2, c = (tid & 3) * 8;
            __builtin_prefetch(&W[(col0 + r) * (long)K + (t + 1) * BK + c], 0, 1);
        }

        Frag16 af[4], bf[2];
        #pragma unroll
        for (int mt = 0; mt < 4; ++mt) {
            int r = waveM * 64 + mt * 16 + lo16;
            af[mt].h[0] = *(const v8bf*)&As[buf][r * LDT + 8 * hl];
            af[mt].h[1] = *(const v8bf*)&As[buf][r * LDT + 16 + 8 * hl];
        }
        #pragma unroll
        for (int nt = 0; nt < 2; ++nt) {
            int r = waveN * 32 + nt * 16 + lo16;
            bf[nt].h[0] = *(const v8bf*)&Bs[buf][r * LDT + 8 * hl];
            bf[nt].h[1] = *(const v8bf*)&Bs[buf][r * LDT + 16 + 8 * hl];
        }
        #pragma unroll
        for (int mt = 0; mt < 4; ++mt)
            #pragma unroll
            for (int nt = 0; nt < 2; ++nt)
                acc[mt][nt] = wmma_bf16(af[mt].v, bf[nt].v, acc[mt][nt]);
        __syncthreads();                  // reads of buf done before t+2 overwrites it
    }

    // epilogue: lane holds C rows m = 8*hl + g, col = lo16 (per 16x16 tile)
    #pragma unroll
    for (int mt = 0; mt < 4; ++mt) {
        #pragma unroll
        for (int nt = 0; nt < 2; ++nt) {
            long col = col0 + waveN * 32 + nt * 16 + lo16;
            float bi = bias[col];
            #pragma unroll
            for (int g = 0; g < 8; ++g) {
                long mrow = row0 + waveM * 64 + mt * 16 + 8 * hl + g;
                float v = acc[mt][nt][g] + bi;
                long idx = mrow * (long)N + col;
                if constexpr (EPI == 0) {
                    ((float*)outv)[idx] = v;
                } else if constexpr (EPI == 1) {
                    ((u16*)outv)[idx] = f2bf(gelu_tanh(v));
                } else if constexpr (EPI == 2) {
                    ((float*)outv)[idx] = res[idx] + v;
                } else {
                    int bb = (int)(mrow / rows_per_batch);
                    float gate = modbuf[((long)bb * 6 + gate_idx) * nDIM + col];
                    ((float*)outv)[idx] = res[idx] + gate * v;
                }
            }
        }
    }
}

// ---------------- flash attention (WMMA) ----------------
// Q,K,V: bf16 [B*H, seq, HD]; out: bf16 merged [B, seq_q, DIM]
__global__ __launch_bounds__(256) void k_attn(const u16* __restrict__ Q,
                                              const u16* __restrict__ Kv,
                                              const u16* __restrict__ Vv,
                                              u16* __restrict__ out,
                                              int seq_q, int seq_kv, float scale) {
    __shared__ u16  Ks[32 * 136];       // [key][d], padded
    __shared__ u16  Vt[128 * 40];       // [d][key], padded
    __shared__ float Sb[8][16 * 36];    // per-wave logits scratch
    __shared__ u16  Pb[8][16 * 40];     // per-wave P (bf16)
    __shared__ float rowM[8][16], rowS[8][16], rowF[8][16];

    const int tid = threadIdx.x;
    const int wave = tid >> 5, lane = tid & 31;
    const int hl = lane >> 4, lo16 = lane & 15;
    const int bh = blockIdx.y;
    const int b = bh / nH, h = bh % nH;
    const long qbase = (long)bh * seq_q * nHD;
    const long kbase = (long)bh * seq_kv * nHD;
    const int q0 = blockIdx.x * 128 + wave * 16;

    // resident Q fragments (16 queries x 128 d -> 4 A-frags)
    Frag16 qf[4];
    {
        const u16* qp = Q + qbase + (long)(q0 + lo16) * nHD;
        #pragma unroll
        for (int dt = 0; dt < 4; ++dt) {
            qf[dt].h[0] = *(const v8bf*)&qp[dt * 32 + 8 * hl];
            qf[dt].h[1] = *(const v8bf*)&qp[dt * 32 + 16 + 8 * hl];
        }
    }
    v8f zero = {};
    v8f acc[8];
    #pragma unroll
    for (int t = 0; t < 8; ++t) acc[t] = zero;
    if (lane < 16) { rowM[wave][lane] = -1e30f; rowS[wave][lane] = 0.f; rowF[wave][lane] = 1.f; }

    for (int kb = 0; kb < seq_kv; kb += 32) {
        __syncthreads();
        // async copy of K block (row-major)
        #pragma unroll
        for (int i = 0; i < 2; ++i) {
            int cid = tid + 256 * i;           // 512 chunks of 8
            int kr = cid >> 4, kc = (cid & 15) * 8;
            async_copy16(&Kv[kbase + (long)(kb + kr) * nHD + kc], &Ks[kr * 136 + kc]);
        }
        // V block transposed through VGPRs (sync path)
        #pragma unroll
        for (int i = 0; i < 2; ++i) {
            int cid = tid + 256 * i;
            int kr = cid >> 4, kc = (cid & 15) * 8;
            v4u vv = *(const v4u*)&Vv[kbase + (long)(kb + kr) * nHD + kc];
            const u16* pv = (const u16*)&vv;
            #pragma unroll
            for (int e = 0; e < 8; ++e) Vt[(kc + e) * 40 + kr] = pv[e];
        }
        async_wait_all();
        __syncthreads();

        // logits: two 16-key subtiles
        #pragma unroll
        for (int ktile = 0; ktile < 2; ++ktile) {
            v8f sacc = zero;
            #pragma unroll
            for (int dt = 0; dt < 4; ++dt) {
                Frag16 kf;
                int krow = ktile * 16 + lo16;
                kf.h[0] = *(const v8bf*)&Ks[krow * 136 + dt * 32 + 8 * hl];
                kf.h[1] = *(const v8bf*)&Ks[krow * 136 + dt * 32 + 16 + 8 * hl];
                sacc = wmma_bf16(qf[dt].v, kf.v, sacc);
            }
            #pragma unroll
            for (int g = 0; g < 8; ++g) {
                int m = 8 * hl + g;
                Sb[wave][m * 36 + ktile * 16 + lo16] = sacc[g] * scale;
            }
        }
        // online softmax: 2 lanes per query row
        {
            int r = lane >> 1, hh = lane & 1;
            float mx = -1e30f;
            #pragma unroll
            for (int c = 0; c < 16; ++c) mx = fmaxf(mx, Sb[wave][r * 36 + hh * 16 + c]);
            mx = fmaxf(mx, __shfl_xor(mx, 1, 32));
            float mold = rowM[wave][r];
            float mnew = fmaxf(mold, mx);
            float sum = 0.f;
            #pragma unroll
            for (int c = 0; c < 16; ++c) {
                float p = __expf(Sb[wave][r * 36 + hh * 16 + c] - mnew);
                sum += p;
                Pb[wave][r * 40 + hh * 16 + c] = f2bf(p);
            }
            sum += __shfl_xor(sum, 1, 32);
            float sf = __expf(mold - mnew);
            if (hh == 0) {
                rowM[wave][r] = mnew;
                rowS[wave][r] = rowS[wave][r] * sf + sum;
                rowF[wave][r] = sf;
            }
        }
        // rescale running accumulators
        float sc8[8];
        #pragma unroll
        for (int g = 0; g < 8; ++g) sc8[g] = rowF[wave][8 * hl + g];
        #pragma unroll
        for (int t = 0; t < 8; ++t)
            #pragma unroll
            for (int g = 0; g < 8; ++g) acc[t][g] *= sc8[g];
        // PV: A = P(16x32), B = V(32keys x 16d) from transposed LDS
        Frag16 pf;
        pf.h[0] = *(const v8bf*)&Pb[wave][lo16 * 40 + 8 * hl];
        pf.h[1] = *(const v8bf*)&Pb[wave][lo16 * 40 + 16 + 8 * hl];
        #pragma unroll
        for (int t = 0; t < 8; ++t) {
            Frag16 vf;
            int drow = t * 16 + lo16;
            vf.h[0] = *(const v8bf*)&Vt[drow * 40 + 8 * hl];
            vf.h[1] = *(const v8bf*)&Vt[drow * 40 + 16 + 8 * hl];
            acc[t] = wmma_bf16(pf.v, vf.v, acc[t]);
        }
    }

    float inv8[8];
    #pragma unroll
    for (int g = 0; g < 8; ++g) inv8[g] = 1.0f / rowS[wave][8 * hl + g];
    #pragma unroll
    for (int t = 0; t < 8; ++t)
        #pragma unroll
        for (int g = 0; g < 8; ++g) {
            long q = q0 + 8 * hl + g;
            long d = t * 16 + lo16;
            long orow = (long)b * seq_q + q;
            out[orow * nDIM + (long)h * nHD + d] = f2bf(acc[t][g] * inv8[g]);
        }
}

// ---------------- host orchestration ----------------
extern "C" void kernel_launch(void* const* d_in, const int* in_sizes, int n_in,
                              void* d_out, int out_size, void* d_ws, size_t ws_size,
                              hipStream_t stream) {
    (void)in_sizes; (void)n_in; (void)out_size; (void)ws_size;
    const float* hs    = (const float*)d_in[0];
    const float* enc   = (const float*)d_in[1];
    const float* temb  = (const float*)d_in[2];
    const float* rcos  = (const float*)d_in[3];
    const float* rsin  = (const float*)d_in[4];
    const float* sst   = (const float*)d_in[5];
    const float* w_qkv = (const float*)d_in[6];   const float* b_qkv = (const float*)d_in[7];
    const float* rms_q1= (const float*)d_in[8];   const float* rms_k1= (const float*)d_in[9];
    const float* w_o1  = (const float*)d_in[10];  const float* b_o1  = (const float*)d_in[11];
    const float* ln2_g = (const float*)d_in[12];  const float* ln2_b = (const float*)d_in[13];
    const float* w_q2  = (const float*)d_in[14];  const float* b_q2  = (const float*)d_in[15];
    const float* w_kv2 = (const float*)d_in[16];  const float* b_kv2 = (const float*)d_in[17];
    const float* rms_q2= (const float*)d_in[18];  const float* rms_k2= (const float*)d_in[19];
    const float* w_o2  = (const float*)d_in[20];  const float* b_o2  = (const float*)d_in[21];
    const float* w_ff1 = (const float*)d_in[22];  const float* b_ff1 = (const float*)d_in[23];
    const float* w_ff2 = (const float*)d_in[24];  const float* b_ff2 = (const float*)d_in[25];
    float* outp = (float*)d_out;
    char*  ws   = (char*)d_ws;

    const long M1 = (long)nB * nS;     // 4096
    const long Mc = (long)nB * nSC;    // 1024

    size_t off = 0;
    auto take = [&](size_t bytes) { size_t o = off; off += (bytes + 255) & ~(size_t)255; return o; };
    size_t oMOD  = take((size_t)nB * 6 * nDIM * 4);
    size_t oWQKV = take((size_t)3 * nDIM * nDIM * 2);
    size_t oWO1  = take((size_t)nDIM * nDIM * 2);
    size_t oWQ2  = take((size_t)nDIM * nDIM * 2);
    size_t oWKV2 = take((size_t)2 * nDIM * nDIM * 2);
    size_t oWO2  = take((size_t)nDIM * nDIM * 2);
    size_t oWFF1 = take((size_t)nFFN * nDIM * 2);
    size_t oWFF2 = take((size_t)nFFN * nDIM * 2);
    size_t oENC  = take((size_t)Mc * nDIM * 2);
    size_t oX    = take((size_t)M1 * nDIM * 2);
    size_t oBIG  = take((size_t)M1 * 3 * nDIM * 4);  // qkv f32 | later {q2,kv} f32 | later ff1 bf16
    size_t oQH   = take((size_t)M1 * nDIM * 2);
    size_t oKH   = take((size_t)M1 * nDIM * 2);
    size_t oVH   = take((size_t)M1 * nDIM * 2);
    size_t oATT  = take((size_t)M1 * nDIM * 2);
    size_t oH1   = take((size_t)M1 * nDIM * 4);
    size_t oH2   = take((size_t)M1 * nDIM * 4);

    float* MOD   = (float*)(ws + oMOD);
    u16* WQKVb   = (u16*)(ws + oWQKV);
    u16* WO1b    = (u16*)(ws + oWO1);
    u16* WQ2b    = (u16*)(ws + oWQ2);
    u16* WKV2b   = (u16*)(ws + oWKV2);
    u16* WO2b    = (u16*)(ws + oWO2);
    u16* WFF1b   = (u16*)(ws + oWFF1);
    u16* WFF2b   = (u16*)(ws + oWFF2);
    u16* ENCb    = (u16*)(ws + oENC);
    u16* Xb      = (u16*)(ws + oX);
    float* QKVf  = (float*)(ws + oBIG);
    float* Q2f   = (float*)(ws + oBIG);                                   // reuse
    float* KVf   = (float*)(ws + oBIG + (size_t)M1 * nDIM * 4);           // reuse
    u16*   Fb    = (u16*)(ws + oBIG);                                     // reuse
    u16* QHb = (u16*)(ws + oQH);
    u16* KHb = (u16*)(ws + oKH);
    u16* VHb = (u16*)(ws + oVH);
    u16* ATTb = (u16*)(ws + oATT);
    float* H1 = (float*)(ws + oH1);
    float* H2 = (float*)(ws + oH2);

    auto blocks = [](long n) { return (unsigned)((n + 255) / 256); };

    // modulation table + bf16 conversions
    k_mod<<<blocks((long)nB * 6 * nDIM), 256, 0, stream>>>(sst, temb, MOD);
    k_cvt<<<blocks((long)3 * nDIM * nDIM), 256, 0, stream>>>(w_qkv, WQKVb, (long)3 * nDIM * nDIM);
    k_cvt<<<blocks((long)nDIM * nDIM),     256, 0, stream>>>(w_o1,  WO1b,  (long)nDIM * nDIM);
    k_cvt<<<blocks((long)nDIM * nDIM),     256, 0, stream>>>(w_q2,  WQ2b,  (long)nDIM * nDIM);
    k_cvt<<<blocks((long)2 * nDIM * nDIM), 256, 0, stream>>>(w_kv2, WKV2b, (long)2 * nDIM * nDIM);
    k_cvt<<<blocks((long)nDIM * nDIM),     256, 0, stream>>>(w_o2,  WO2b,  (long)nDIM * nDIM);
    k_cvt<<<blocks((long)nFFN * nDIM),     256, 0, stream>>>(w_ff1, WFF1b, (long)nFFN * nDIM);
    k_cvt<<<blocks((long)nFFN * nDIM),     256, 0, stream>>>(w_ff2, WFF2b, (long)nFFN * nDIM);
    k_cvt<<<blocks(Mc * nDIM),             256, 0, stream>>>(enc,   ENCb,  Mc * nDIM);

    // x = ln(h)*(1+scale_sa)+shift_sa  (shift=idx0, scale=idx1)
    k_ln<<<(unsigned)M1, 256, 0, stream>>>(hs, Xb, MOD, 0, 1, nullptr, nullptr, nS, 0);
    // qkv = x @ w_qkv^T + b_qkv
    k_gemm<0><<<dim3(3 * nDIM / 128, (unsigned)(M1 / 128)), 256, 0, stream>>>(
        Xb, WQKVb, b_qkv, QKVf, nullptr, nullptr, 0, 0, (int)M1, 3 * nDIM, nDIM);
    // q,k: rms + rope -> heads ; v: pack
    k_rmsrope<<<(unsigned)M1, 256, 0, stream>>>(QKVf,            3 * nDIM, rms_q1, rcos, rsin, QHb, nS, 1);
    k_rmsrope<<<(unsigned)M1, 256, 0, stream>>>(QKVf + nDIM,     3 * nDIM, rms_k1, rcos, rsin, KHb, nS, 1);
    k_pack   <<<(unsigned)M1, 256, 0, stream>>>(QKVf + 2 * nDIM, 3 * nDIM, VHb, nS);
    // self attention
    k_attn<<<dim3(nS / 128, nB * nH), 256, 0, stream>>>(QHb, KHb, VHb, ATTb, nS, nS, kAttnScale);
    // h1 = h + gate_sa * (attn @ w_o1^T + b_o1)
    k_gemm<3><<<dim3(nDIM / 128, (unsigned)(M1 / 128)), 256, 0, stream>>>(
        ATTb, WO1b, b_o1, H1, hs, MOD, 2, nS, (int)M1, nDIM, nDIM);
    // x = ln(h1)*g + b
    k_ln<<<(unsigned)M1, 256, 0, stream>>>(H1, Xb, nullptr, 0, 0, ln2_g, ln2_b, nS, 1);
    // q2 = x @ w_q2^T ; kv = enc @ w_kv2^T
    k_gemm<0><<<dim3(nDIM / 128, (unsigned)(M1 / 128)), 256, 0, stream>>>(
        Xb, WQ2b, b_q2, Q2f, nullptr, nullptr, 0, 0, (int)M1, nDIM, nDIM);
    k_gemm<0><<<dim3(2 * nDIM / 128, (unsigned)(Mc / 128)), 256, 0, stream>>>(
        ENCb, WKV2b, b_kv2, KVf, nullptr, nullptr, 0, 0, (int)Mc, 2 * nDIM, nDIM);
    // rms (no rope) + pack heads
    k_rmsrope<<<(unsigned)M1, 256, 0, stream>>>(Q2f,        nDIM,     rms_q2, rcos, rsin, QHb, nS, 0);
    k_rmsrope<<<(unsigned)Mc, 256, 0, stream>>>(KVf,        2 * nDIM, rms_k2, rcos, rsin, KHb, nSC, 0);
    k_pack   <<<(unsigned)Mc, 256, 0, stream>>>(KVf + nDIM, 2 * nDIM, VHb, nSC);
    // cross attention
    k_attn<<<dim3(nS / 128, nB * nH), 256, 0, stream>>>(QHb, KHb, VHb, ATTb, nS, nSC, kAttnScale);
    // h2 = h1 + attn @ w_o2^T + b_o2
    k_gemm<2><<<dim3(nDIM / 128, (unsigned)(M1 / 128)), 256, 0, stream>>>(
        ATTb, WO2b, b_o2, H2, H1, nullptr, 0, 0, (int)M1, nDIM, nDIM);
    // x = ln(h2)*(1+scale_ff)+shift_ff  (shift=idx3, scale=idx4)
    k_ln<<<(unsigned)M1, 256, 0, stream>>>(H2, Xb, MOD, 3, 4, nullptr, nullptr, nS, 0);
    // f = gelu(x @ w_ff1^T + b_ff1)  (bf16)
    k_gemm<1><<<dim3(nFFN / 128, (unsigned)(M1 / 128)), 256, 0, stream>>>(
        Xb, WFF1b, b_ff1, Fb, nullptr, nullptr, 0, 0, (int)M1, nFFN, nDIM);
    // out = h2 + gate_ff * (f @ w_ff2^T + b_ff2)
    k_gemm<3><<<dim3(nDIM / 128, (unsigned)(M1 / 128)), 256, 0, stream>>>(
        Fb, WFF2b, b_ff2, outp, H2, MOD, 5, nS, (int)M1, nDIM, nFFN);
}
